// ModulatedStyleConv_63367947485949
// MI455X (gfx1250) — matrix operand order repaired and checked by reading, not compiled, not run on hardware
//
#include <hip/hip_runtime.h>

// ---------------------------------------------------------------------------
// ModulatedStyleConv for MI455X (gfx1250):
//   s = style @ mod_weight^T + mod_bias
//   demod[n,co] = rsqrt( sum_ci s[n,ci]^2 * Q[co,ci] + eps ),  Q = sum_k W^2
//   out = leaky( demod[n,co]*sqrt2 * conv(W, s[n,ci]*x) + nw*noise + bias )
// Conv = implicit GEMM on v_wmma_f32_16x16x32_bf16 (wave32), with
// double-buffered global_load_async_to_lds input staging.
// ---------------------------------------------------------------------------

#define N_      8
#define CIN_    512
#define COUT_   512
#define H_      64
#define W_      64
#define HP_     66   // padded
#define EPS_    1e-8f
#define SQRT2_  1.41421356237309515f

typedef __attribute__((ext_vector_type(16))) __bf16 v16bf;
typedef __attribute__((ext_vector_type(8)))  __bf16 v8bf;
typedef __attribute__((ext_vector_type(8)))  float  v8f;
typedef __attribute__((ext_vector_type(4)))  int    v4i;

#if defined(__AMDGCN__) && __has_builtin(__builtin_amdgcn_global_load_async_to_lds_b128)
#define HAVE_ASYNC 1
#else
#define HAVE_ASYNC 0
#endif

__device__ __forceinline__ void async_cp_b128(const unsigned short* g, unsigned short* l) {
#if HAVE_ASYNC
    __builtin_amdgcn_global_load_async_to_lds_b128(
        (__attribute__((address_space(1))) v4i*)(const_cast<unsigned short*>(g)),
        (__attribute__((address_space(3))) v4i*)l, 0, 0);
#else
    *(uint4*)l = *(const uint4*)g;
#endif
}

__device__ __forceinline__ void wait_async() {
#if HAVE_ASYNC
#if __has_builtin(__builtin_amdgcn_s_wait_asynccnt)
    __builtin_amdgcn_s_wait_asynccnt(0);
#else
    asm volatile("s_wait_asynccnt 0" ::: "memory");
#endif
#endif
}

// round-to-nearest-even float -> bf16 bits
__device__ __forceinline__ unsigned short f2bf(float f) {
    unsigned u = __float_as_uint(f);
    unsigned r = u + 0x7FFFu + ((u >> 16) & 1u);
    return (unsigned short)(r >> 16);
}

// --------------------------- prep kernels ----------------------------------

// s[n,ci] = sum_j style[n,j] * mod_weight[ci,j] + mod_bias[ci]   (8x512, K=512)
__global__ __launch_bounds__(256) void k_style_mod(
    const float* __restrict__ style, const float* __restrict__ mw,
    const float* __restrict__ mb, float* __restrict__ s)
{
    int tid = blockIdx.x * 256 + threadIdx.x;     // 4096
    int n = tid >> 9, ci = tid & 511;
    const float* sr = style + n * CIN_;
    const float* wr = mw + (size_t)ci * CIN_;
    float acc = mb[ci];
    #pragma unroll 8
    for (int j = 0; j < CIN_; ++j) acc += sr[j] * wr[j];
    s[tid] = acc;
}

// Q[co,ci] = sum over 9 taps of conv_weight^2
__global__ __launch_bounds__(256) void k_qsum(
    const float* __restrict__ cw, float* __restrict__ q)
{
    int tid = blockIdx.x * 256 + threadIdx.x;     // 262144
    const float* p = cw + (size_t)tid * 9;
    float acc = 0.f;
    #pragma unroll
    for (int k = 0; k < 9; ++k) acc += p[k] * p[k];
    q[tid] = acc;
}

// demod[n,co] = rsqrt( sum_ci s[n,ci]^2 * Q[co,ci] + eps )
__global__ __launch_bounds__(256) void k_demod(
    const float* __restrict__ s, const float* __restrict__ q,
    float* __restrict__ dm)
{
    int tid = blockIdx.x * 256 + threadIdx.x;     // 4096
    int n = tid >> 9, co = tid & 511;
    const float* sr = s + n * CIN_;
    const float* qr = q + (size_t)co * CIN_;
    float acc = 0.f;
    #pragma unroll 8
    for (int ci = 0; ci < CIN_; ++ci) { float sv = sr[ci]; acc += sv * sv * qr[ci]; }
    dm[tid] = rsqrtf(acc + EPS_);
}

// wb[(tap*COUT + co)*CIN + ci] = bf16(conv_weight[co,ci,kh,kw]), tap = kh*3+kw
__global__ __launch_bounds__(256) void k_wcvt(
    const float* __restrict__ cw, unsigned short* __restrict__ wb)
{
    int tid = blockIdx.x * 256 + threadIdx.x;     // 2359296
    int kk = tid % 9;
    int rc = tid / 9;                             // co*512 + ci
    int ci = rc & 511, co = rc >> 9;
    wb[((size_t)kk * COUT_ + co) * CIN_ + ci] = f2bf(cw[tid]);
}

// xsp[n][h'][x'][ci] = (interior) ? bf16( x[n,ci,h'-1,x'-1] * s[n,ci] ) : 0
__global__ __launch_bounds__(256) void k_xprep(
    const float* __restrict__ x, const float* __restrict__ s,
    unsigned short* __restrict__ xsp)
{
    int tid = blockIdx.x * 256 + threadIdx.x;     // 8*66*66*512
    int ci = tid & 511;
    int rest = tid >> 9;
    int xx = rest % HP_; rest /= HP_;
    int hh = rest % HP_;
    int n  = rest / HP_;
    float v = 0.f;
    if (hh >= 1 && hh <= H_ && xx >= 1 && xx <= W_) {
        v = x[(((size_t)n * CIN_ + ci) * H_ + (hh - 1)) * W_ + (xx - 1)] * s[n * CIN_ + ci];
    }
    xsp[tid] = f2bf(v);
}

// --------------------------- WMMA conv kernel ------------------------------
// grid: (COUT/256, H, N); block: 256 = 8 waves. Wave w owns couts w*32..w*32+31
// (two 16-row M tiles), 64 pixels of row h (four 16-px N tiles).
// Weights (A-frags) come straight from L2 (layout matches ISA A striping);
// input row slice is double-buffered in LDS via async-to-LDS.
__global__ __launch_bounds__(256) void k_conv_wmma(
    const unsigned short* __restrict__ xsp,   // [N][66][66][512] bf16
    const unsigned short* __restrict__ wb,    // [9][COUT][CIN] bf16
    const float* __restrict__ dm,             // [N][COUT]
    const float* __restrict__ noise,          // [N][64][64]
    const float* __restrict__ nw,             // [1]
    const float* __restrict__ abias,          // [COUT]
    float* __restrict__ out)                  // [N][COUT][64][64]
{
    __shared__ alignas(16) unsigned short sm[2 * 2112];   // 2 x [66 px][32 cin]

    const int tid  = threadIdx.x;
    const int lane = tid & 31, wave = tid >> 5;
    const int lm = lane & 15, hi = lane >> 4;
    const int coutBase = blockIdx.x * 256;
    const int h = blockIdx.y;
    const int n = blockIdx.z;

    v8f acc[2][4];
    #pragma unroll
    for (int m = 0; m < 2; ++m)
        #pragma unroll
        for (int t = 0; t < 4; ++t) acc[m][t] = (v8f){};

    const size_t xrow = (((size_t)n * HP_ + h) * HP_) * CIN_;   // + dh*66*512

    // stage input slice for step (cb = step/3, dh = step%3) into buffer `buf`
    auto stage = [&](int step, int buf) {
        const int cb = step / 3, dh = step - cb * 3;
        const size_t gbase = xrow + (size_t)dh * (HP_ * CIN_) + (size_t)cb * 32;
        unsigned short* lb = sm + buf * 2112;
        {   // chunks 0..255 (16B each)
            int c = tid, xx = c >> 2, sub = c & 3;
            async_cp_b128(xsp + gbase + (size_t)xx * CIN_ + sub * 8, lb + c * 8);
        }
        if (tid < 8) {  // chunks 256..263
            int c = 256 + tid, xx = c >> 2, sub = c & 3;
            async_cp_b128(xsp + gbase + (size_t)xx * CIN_ + sub * 8, lb + c * 8);
        }
    };

    stage(0, 0);
    int cur = 0;
    for (int step = 0; step < 48; ++step) {
        wait_async();
        __syncthreads();                       // buf[cur] ready for all waves
        if (step + 1 < 48) stage(step + 1, cur ^ 1);

        const int cb = step / 3, dh = step - cb * 3;
        const unsigned short* lb = sm + cur * 2112;

        #pragma unroll
        for (int dw = 0; dw < 3; ++dw) {
            // A frags: two 16(cout) x 32(cin) tiles, direct from L2
            v16bf a[2];
            #pragma unroll
            for (int m = 0; m < 2; ++m) {
                size_t g = ((size_t)(dh * 3 + dw) * COUT_
                            + coutBase + wave * 32 + m * 16 + lm) * CIN_
                         + cb * 32 + hi * 8;
                ((v8bf*)&a[m])[0] = *(const v8bf*)(wb + g);
                ((v8bf*)&a[m])[1] = *(const v8bf*)(wb + g + 16);
            }
            #pragma unroll
            for (int t = 0; t < 4; ++t) {
                // B frag: 32(cin) x 16(pixels), shifted by dw
                int p = t * 16 + lm + dw;
                int off = p * 32 + hi * 8;
                v16bf b;
                ((v8bf*)&b)[0] = *(const v8bf*)(lb + off);
                ((v8bf*)&b)[1] = *(const v8bf*)(lb + off + 16);
                acc[0][t] = __builtin_amdgcn_wmma_f32_16x16x32_bf16(
                    false, a[0], false, b, (short)0, acc[0][t], false, false);
                acc[1][t] = __builtin_amdgcn_wmma_f32_16x16x32_bf16(
                    false, a[1], false, b, (short)0, acc[1][t], false, false);
            }
        }
        cur ^= 1;
    }

    // epilogue: *demod*sqrt2 + noise + bias, leaky relu(0.2)
    const float nw0 = nw[0];
    #pragma unroll
    for (int m = 0; m < 2; ++m) {
        float dsc[8], bs[8];
        #pragma unroll
        for (int r = 0; r < 8; ++r) {
            int co = coutBase + wave * 32 + m * 16 + r + hi * 8;
            dsc[r] = dm[n * COUT_ + co] * SQRT2_;
            bs[r]  = abias[co];
        }
        #pragma unroll
        for (int t = 0; t < 4; ++t) {
            int w = t * 16 + lm;
            float nz = nw0 * noise[((size_t)n * H_ + h) * W_ + w];
            #pragma unroll
            for (int r = 0; r < 8; ++r) {
                int co = coutBase + wave * 32 + m * 16 + r + hi * 8;
                float v = acc[m][t][r] * dsc[r] + nz + bs[r];
                v = v >= 0.f ? v : 0.2f * v;
                out[(((size_t)n * COUT_ + co) * H_ + h) * W_ + w] = v;
            }
        }
    }
}

// --------------------------- launcher --------------------------------------

extern "C" void kernel_launch(void* const* d_in, const int* in_sizes, int n_in,
                              void* d_out, int out_size, void* d_ws, size_t ws_size,
                              hipStream_t stream) {
    const float* x      = (const float*)d_in[0];
    const float* style  = (const float*)d_in[1];
    const float* noise  = (const float*)d_in[2];
    const float* cw     = (const float*)d_in[3];
    const float* mw     = (const float*)d_in[4];
    const float* mb     = (const float*)d_in[5];
    const float* nw     = (const float*)d_in[6];
    const float* abias  = (const float*)d_in[7];
    float* out = (float*)d_out;

    char* ws = (char*)d_ws;
    float*          s_buf = (float*)(ws + 0);                  // 16 KB
    float*          dm    = (float*)(ws + 16384);              // 16 KB
    float*          q     = (float*)(ws + 32768);              // 1 MB
    unsigned short* wb    = (unsigned short*)(ws + 1081344);   // 4.5 MB
    unsigned short* xsp   = (unsigned short*)(ws + 5799936);   // 35.7 MB

    k_style_mod<<<16, 256, 0, stream>>>(style, mw, mb, s_buf);
    k_qsum<<<1024, 256, 0, stream>>>(cw, q);
    k_demod<<<16, 256, 0, stream>>>(s_buf, q, dm);
    k_wcvt<<<9216, 256, 0, stream>>>(cw, wb);
    k_xprep<<<69696, 256, 0, stream>>>(x, s_buf, xsp);
    k_conv_wmma<<<dim3(COUT_ / 256, H_, N_), 256, 0, stream>>>(
        xsp, wb, dm, noise, nw, abias, out);
}
